// DANetHead_36953898614884
// MI455X (gfx1250) — matrix operand order repaired and checked by reading, not compiled, not run on hardware
//
#include <hip/hip_runtime.h>
#include <hip/hip_bf16.h>

#define Bn 8
#define Cc 256
#define Nn 4096
#define Pp 64
#define Hh 8
#define Dd 32

typedef _Float16 h16 __attribute__((ext_vector_type(16)));
typedef _Float16 h8  __attribute__((ext_vector_type(8)));
typedef float    f8  __attribute__((ext_vector_type(8)));

__device__ __forceinline__ f8 f8zero() {
  f8 z = {0.f, 0.f, 0.f, 0.f, 0.f, 0.f, 0.f, 0.f};
  return z;
}

__device__ __forceinline__ h16 cat8(h8 lo, h8 hi) {
  return __builtin_shufflevector(lo, hi, 0,1,2,3,4,5,6,7,8,9,10,11,12,13,14,15);
}

// A-operand (16x32 f16, M x K): lane m(0-15) holds row m, K {0..7,16..23};
// lanes 16-31 hold row m, K {8..15,24..31}. Rows contiguous-K with stride ld.
__device__ __forceinline__ h16 load_a(const _Float16* base, int ld, int lane) {
  int m = lane & 15, s = (lane >> 4) * 8;
  const _Float16* p = base + (long)m * ld + s;
  h8 lo = *(const h8*)p;
  h8 hi = *(const h8*)(p + 16);
  return cat8(lo, hi);
}

// B-operand (32x16 f16, K x N): lane n(0-15) holds col n, K {0..7,16..23}; lanes 16-31
// hold col n, K {8..15,24..31}. Storage: bstore[n][k] (k contiguous), stride ld.
__device__ __forceinline__ h16 load_b(const _Float16* base, int ld, int lane) {
  int n = lane & 15, s = (lane >> 4) * 8;
  const _Float16* p = base + (long)n * ld + s;
  h8 lo = *(const h8*)p;
  h8 hi = *(const h8*)(p + 16);
  return cat8(lo, hi);
}

__device__ __forceinline__ f8 wmma16(h16 a, h16 b, f8 c) {
  return __builtin_amdgcn_wmma_f32_16x16x32_f16(false, a, false, b, (short)0, c,
                                                false, false);
}

// ---- CDNA5 async global->LDS copy (ASYNCcnt-tracked, per-lane 16B) ----
__device__ __forceinline__ void async_ld_b128(void* lds_ptr, const void* gaddr) {
  unsigned int lds_off = (unsigned int)(unsigned long)lds_ptr;  // addr[31:0] = LDS offset
  asm volatile("global_load_async_to_lds_b128 %0, %1, off"
               :
               : "v"(lds_off), "v"(gaddr)
               : "memory");
}
__device__ __forceinline__ void wait_async0() {
  asm volatile("s_wait_asynccnt 0" ::: "memory");
}

// -------------------- stage 1: transpose / convert --------------------
// dst[c][r] = src[r][c], f32 -> f16, per batch
__global__ void k_transpose_f32_f16(_Float16* __restrict__ dst,
                                    const float* __restrict__ src,
                                    int rows, int cols,
                                    long srcBatch, long dstBatch) {
  __shared__ float tile[32][33];
  int b = blockIdx.z;
  int r0 = blockIdx.x * 32, c0 = blockIdx.y * 32;
  const float* s = src + (long)b * srcBatch;
  _Float16* d = dst + (long)b * dstBatch;
  int tx = threadIdx.x, ty = threadIdx.y;  // (32,8)
#pragma unroll
  for (int j = 0; j < 4; ++j) {
    int r = ty + j * 8;
    tile[r][tx] = s[(long)(r0 + r) * cols + (c0 + tx)];
  }
  __syncthreads();
#pragma unroll
  for (int j = 0; j < 4; ++j) {
    int r = ty + j * 8;
    d[(long)(c0 + r) * rows + (r0 + tx)] = (_Float16)tile[tx][r];
  }
}

__global__ void k_cvt_f16(_Float16* __restrict__ dst, const float* __restrict__ src, int n) {
  int i = blockIdx.x * 256 + threadIdx.x;
  if (i < n) dst[i] = (_Float16)src[i];
}

// -------------------- stage 2: QKV GEMM (WMMA) --------------------
// y = xf(N x C) @ W^T, block tile 128(M) x 64(cols), K=C=256.
// Whole 64x256 W panel is async-copied to LDS once; K loop runs barrier-free.
__global__ void __launch_bounds__(256)
k_qkv_gemm(const _Float16* __restrict__ XH,
           const _Float16* __restrict__ WSAH, const _Float16* __restrict__ WSCH,
           _Float16* __restrict__ QSA, _Float16* __restrict__ KSAT,
           _Float16* __restrict__ VSAT, _Float16* __restrict__ QCT,
           _Float16* __restrict__ KCT, _Float16* __restrict__ VC) {
  __shared__ _Float16 wt[8 * 64 * 40];  // [kslab][oc][40] padded (conflict-free b128)
  int z = blockIdx.z;
  int b = z >> 1, mode = z & 1;
  const _Float16* W = mode ? WSCH : WSAH;
  int m0 = blockIdx.x * 128;
  int cb = blockIdx.y;            // 0..11
  int chunk = cb >> 2;            // which 256-col chunk (q/k/v)
  int ocBase = (cb & 3) * 64;     // col offset within chunk
  int wrowBase = (mode == 0 && chunk == 2) ? 3 * Cc : chunk * Cc;
  int t = threadIdx.x, lane = t & 31, wv = t >> 5;

  // async preload: 64 rows x 256 halves, split into 8 K-slabs of 32 halves
  {
    int oc = t >> 2, part = (t & 3) * 8;
    const _Float16* wrow = W + (long)(wrowBase + ocBase + oc) * Cc + part;
#pragma unroll
    for (int s = 0; s < 8; ++s)
      async_ld_b128(&wt[(s * 64 + oc) * 40 + part], wrow + s * 32);
  }
  wait_async0();
  __syncthreads();

  f8 acc[4];
#pragma unroll
  for (int j = 0; j < 4; ++j) acc[j] = f8zero();

  const _Float16* Abase = XH + ((long)b * Nn + m0 + wv * 16) * Cc;

#pragma unroll
  for (int s = 0; s < 8; ++s) {
    h16 a = load_a(Abase + s * 32, Cc, lane);
    const _Float16* slab = &wt[s * 64 * 40];
    h16 b0 = load_b(slab + 0 * 16 * 40, 40, lane);
    h16 b1 = load_b(slab + 1 * 16 * 40, 40, lane);
    h16 b2 = load_b(slab + 2 * 16 * 40, 40, lane);
    h16 b3 = load_b(slab + 3 * 16 * 40, 40, lane);
    acc[0] = wmma16(a, b0, acc[0]);
    acc[1] = wmma16(a, b1, acc[1]);
    acc[2] = wmma16(a, b2, acc[2]);
    acc[3] = wmma16(a, b3, acc[3]);
  }

  int nc = lane & 15, hs = lane >> 4;
#pragma unroll
  for (int j = 0; j < 4; ++j) {
    int oc = ocBase + j * 16 + nc;      // 0..255 within chunk
    int h = oc >> 5, dd = oc & 31;
    long bh = (long)b * Hh + h;
#pragma unroll
    for (int r = 0; r < 8; ++r) {
      int n = m0 + wv * 16 + r + 8 * hs;
      _Float16 hv = (_Float16)acc[j][r];
      if (mode == 0) {
        if (chunk == 0)      QSA [(bh * Nn + n) * Dd + dd] = hv;   // (N,d)
        else if (chunk == 1) KSAT[(bh * Dd + dd) * Nn + n] = hv;   // (d,N)
        else                 VSAT[(bh * Dd + dd) * Nn + n] = hv;   // (d,N)
      } else {
        if (chunk == 0)      QCT [(bh * Dd + dd) * Nn + n] = hv;   // (d,N)
        else if (chunk == 1) KCT [(bh * Dd + dd) * Nn + n] = hv;   // (d,N)
        else                 VC  [(bh * Nn + n) * Dd + dd] = hv;   // (N,d)
      }
    }
  }
}

// -------------------- stage 3: L2 norms (1/max(||.||,eps)) --------------------
__global__ void __launch_bounds__(256)
k_norm_nd(const _Float16* __restrict__ Q, float* __restrict__ INV) {
  __shared__ float red[256];
  int bh = blockIdx.x, t = threadIdx.x;
  int dd = t & 31, seg = t >> 5;
  const _Float16* p = Q + (long)bh * Nn * Dd;
  float s = 0.f;
  for (int n = seg * 512; n < seg * 512 + 512; ++n) {
    float v = (float)p[(long)n * Dd + dd];
    s += v * v;
  }
  red[t] = s;
  __syncthreads();
  if (t < 32) {
    float tot = 0.f;
#pragma unroll
    for (int j = 0; j < 8; ++j) tot += red[j * 32 + t];
    INV[bh * 32 + t] = 1.0f / fmaxf(sqrtf(tot), 1e-12f);
  }
}

__global__ void __launch_bounds__(256)
k_norm_dn(const _Float16* __restrict__ A, const _Float16* __restrict__ B2,
          float* __restrict__ IA, float* __restrict__ IB) {
  __shared__ float red[256];
  int bh = blockIdx.x, sel = blockIdx.y, t = threadIdx.x;
  const _Float16* S = sel ? B2 : A;
  int dd = t >> 3, seg = t & 7;
  const _Float16* p = S + ((long)bh * Dd + dd) * Nn;
  float s = 0.f;
  for (int n = seg * 512; n < seg * 512 + 512; ++n) {
    float v = (float)p[n];
    s += v * v;
  }
  red[t] = s;
  __syncthreads();
  if (t < 32) {
    float tot = 0.f;
#pragma unroll
    for (int j = 0; j < 8; ++j) tot += red[t * 8 + j];
    (sel ? IB : IA)[bh * 32 + t] = 1.0f / fmaxf(sqrtf(tot), 1e-12f);
  }
}

// -------------------- stage 4: Linformer projections k_proj/v_proj --------------------
// (d x N) @ EF(N x P): M=32, N'=64, K=4096. 8 waves = 2 Mtiles x 4 Ptiles.
__global__ void __launch_bounds__(256)
k_proj(const _Float16* __restrict__ KSAT, const _Float16* __restrict__ VSAT,
       const _Float16* __restrict__ EFT, _Float16* __restrict__ KPT,
       _Float16* __restrict__ VP) {
  int bh = blockIdx.x, sel = blockIdx.y;
  int t = threadIdx.x, lane = t & 31, wv = t >> 5;
  int mtile = wv >> 2, ptile = wv & 3;
  const _Float16* src = sel ? VSAT : KSAT;
  const _Float16* Ab = src + ((long)bh * Dd + mtile * 16) * Nn;
  const _Float16* Bb = EFT + (long)(ptile * 16) * Nn;
  f8 acc = f8zero();
  for (int kk = 0; kk < Nn; kk += 32)
    acc = wmma16(load_a(Ab + kk, Nn, lane), load_b(Bb + kk, Nn, lane), acc);
  int nc = lane & 15, hs = lane >> 4;
#pragma unroll
  for (int r = 0; r < 8; ++r) {
    int d = mtile * 16 + r + 8 * hs;
    int p = ptile * 16 + nc;
    _Float16 v = (_Float16)acc[r];
    if (sel) VP [(long)bh * (Dd * Pp) + (long)d * Pp + p] = v;  // (d,P)
    else     KPT[(long)bh * (Pp * Dd) + (long)p * Dd + d] = v;  // (P,d)
  }
}

// -------------------- stage 5: channel Gram matrix qc·kc^T --------------------
__global__ void __launch_bounds__(128)
k_ca_gemm(const _Float16* __restrict__ QCT, const _Float16* __restrict__ KCT,
          float* __restrict__ ACA) {
  int bh = blockIdx.x;
  int t = threadIdx.x, lane = t & 31, wv = t >> 5;  // 4 waves
  int mtile = wv >> 1, etile = wv & 1;
  const _Float16* Ab = QCT + ((long)bh * Dd + mtile * 16) * Nn;
  const _Float16* Bb = KCT + ((long)bh * Dd + etile * 16) * Nn;
  f8 acc = f8zero();
  for (int kk = 0; kk < Nn; kk += 32)
    acc = wmma16(load_a(Ab + kk, Nn, lane), load_b(Bb + kk, Nn, lane), acc);
  int nc = lane & 15, hs = lane >> 4;
#pragma unroll
  for (int r = 0; r < 8; ++r) {
    int d = mtile * 16 + r + 8 * hs, e = etile * 16 + nc;
    ACA[(long)bh * 1024 + d * 32 + e] = acc[r];
  }
}

// scale by inv-norms + temp, softmax over e, emit f16 A-operand matrix
__global__ void __launch_bounds__(32)
k_ca_softmax(const float* __restrict__ ACA, const float* __restrict__ IQ,
             const float* __restrict__ IK, const float* __restrict__ temp,
             _Float16* __restrict__ ACAH) {
  int bh = blockIdx.x, h = bh & 7, d = threadIdx.x;
  const float* row = ACA + (long)bh * 1024 + d * 32;
  float iq = IQ[bh * 32 + d], tc = temp[h];
  float v[32];
  float mx = -3.0e38f;
#pragma unroll
  for (int e = 0; e < 32; ++e) {
    v[e] = row[e] * iq * IK[bh * 32 + e] * tc;
    mx = fmaxf(mx, v[e]);
  }
  float sum = 0.f;
#pragma unroll
  for (int e = 0; e < 32; ++e) { v[e] = __expf(v[e] - mx); sum += v[e]; }
  float rinv = 1.0f / sum;
#pragma unroll
  for (int e = 0; e < 32; ++e)
    ACAH[(long)bh * 1024 + d * 32 + e] = (_Float16)(v[e] * rinv);
}

// -------------------- stage 6: fused spatial attention --------------------
// per wave: 16 rows of n. logits = q_n(16x32)@kp^T -> softmax(P=64) -> @v_proj.
__global__ void __launch_bounds__(256)
k_spatial(const _Float16* __restrict__ QSA, const _Float16* __restrict__ KPT,
          const _Float16* __restrict__ VP, const float* __restrict__ INVQ,
          const float* __restrict__ temp2, float* __restrict__ OUT) {
  __shared__ _Float16 kps[64 * 40];      // kp_t scaled by invq[d], [p][d]
  __shared__ _Float16 vps[32 * 72];      // v_proj [d][p]
  __shared__ _Float16 at[8 * 16 * 72];   // per-wave normalized probs (A-operand)
  int bh = blockIdx.x, b = bh >> 3, h = bh & 7;
  int n0 = blockIdx.y * 128;
  int t = threadIdx.x, lane = t & 31, wv = t >> 5;

  // vps: plain copy -> async to LDS; kps: scaled copy (VALU) in parallel
  {
    int d2 = t >> 3, p2 = (t & 7) * 8;
    async_ld_b128(&vps[d2 * 72 + p2], VP + (long)bh * 2048 + t * 8);
  }
#pragma unroll
  for (int i = 0; i < 8; ++i) {
    int e = t * 8 + i;
    int p = e >> 5, d = e & 31;
    kps[p * 40 + d] = (_Float16)((float)KPT[(long)bh * 2048 + e] * INVQ[bh * 32 + d]);
  }
  wait_async0();
  __syncthreads();

  float t2 = temp2[h];
  const _Float16* Ab = QSA + ((long)bh * Nn + n0 + wv * 16) * Dd;
  h16 a = load_a(Ab, Dd, lane);
  h16 kb0 = load_b(&kps[0 * 16 * 40], 40, lane);
  h16 kb1 = load_b(&kps[1 * 16 * 40], 40, lane);
  h16 kb2 = load_b(&kps[2 * 16 * 40], 40, lane);
  h16 kb3 = load_b(&kps[3 * 16 * 40], 40, lane);
  f8 acc[4];
  acc[0] = wmma16(a, kb0, f8zero());
  acc[1] = wmma16(a, kb1, f8zero());
  acc[2] = wmma16(a, kb2, f8zero());
  acc[3] = wmma16(a, kb3, f8zero());

  _Float16* myat = &at[wv * 16 * 72];
  int nc = lane & 15, hs = lane >> 4;
#pragma unroll
  for (int r = 0; r < 8; ++r) {
    float x0 = acc[0][r] * t2, x1 = acc[1][r] * t2;
    float x2 = acc[2][r] * t2, x3 = acc[3][r] * t2;
    float mx = fmaxf(fmaxf(x0, x1), fmaxf(x2, x3));
#pragma unroll
    for (int off = 8; off >= 1; off >>= 1) mx = fmaxf(mx, __shfl_xor(mx, off, 16));
    x0 = __expf(x0 - mx); x1 = __expf(x1 - mx);
    x2 = __expf(x2 - mx); x3 = __expf(x3 - mx);
    float s = x0 + x1 + x2 + x3;
#pragma unroll
    for (int off = 8; off >= 1; off >>= 1) s += __shfl_xor(s, off, 16);
    float ri = 1.0f / s;
    int m = r + 8 * hs;
    myat[m * 72 +  0 + nc] = (_Float16)(x0 * ri);
    myat[m * 72 + 16 + nc] = (_Float16)(x1 * ri);
    myat[m * 72 + 32 + nc] = (_Float16)(x2 * ri);
    myat[m * 72 + 48 + nc] = (_Float16)(x3 * ri);
  }
  __syncthreads();

  f8 o0 = f8zero(), o1 = f8zero();
#pragma unroll
  for (int s2 = 0; s2 < 2; ++s2) {
    h16 aa = load_a(myat + s2 * 32, 72, lane);
    h16 vb0 = load_b(&vps[0 * 72 + s2 * 32], 72, lane);
    h16 vb1 = load_b(&vps[16 * 72 + s2 * 32], 72, lane);
    o0 = wmma16(aa, vb0, o0);
    o1 = wmma16(aa, vb1, o1);
  }
  // scatter through the reference's permute(0,3,1,2).reshape(B,N,C) then (B,C,N)
#pragma unroll
  for (int tt = 0; tt < 2; ++tt) {
    f8& o = tt ? o1 : o0;
#pragma unroll
    for (int r = 0; r < 8; ++r) {
      int dd = tt * 16 + nc;
      int nn = n0 + wv * 16 + r + 8 * hs;
      long flat = (long)dd * ((long)Hh * Nn) + (long)h * Nn + nn;
      int np = (int)(flat >> 8), cp = (int)(flat & 255);
      OUT[((long)b * Cc + cp) * Nn + np] = o[r];  // bijective: full init of OUT
    }
  }
}

// -------------------- stage 7: channel attention output, += --------------------
__global__ void __launch_bounds__(256)
k_channel(const _Float16* __restrict__ ACAH, const _Float16* __restrict__ VC,
          float* __restrict__ OUT) {
  int bh = blockIdx.x, b = bh >> 3, h = bh & 7;
  int n0 = blockIdx.y * 64;
  int t = threadIdx.x, lane = t & 31, wv = t >> 5;
  int mtile = wv & 1, ntile = wv >> 1;
  const _Float16* Ab = ACAH + (long)bh * 1024 + mtile * 16 * 32;
  const _Float16* Bb = VC + ((long)bh * Nn + n0 + ntile * 16) * Dd;
  f8 o = wmma16(load_a(Ab, 32, lane), load_b(Bb, 32, lane), f8zero());
  int nc = lane & 15, hs = lane >> 4;
#pragma unroll
  for (int r = 0; r < 8; ++r) {
    int dd = mtile * 16 + r + 8 * hs;
    int n = n0 + ntile * 16 + nc;
    long idx = ((long)b * Cc + h * Dd + dd) * (long)Nn + n;
    OUT[idx] += o[r];
  }
}

// -------------------- launcher --------------------
extern "C" void kernel_launch(void* const* d_in, const int* in_sizes, int n_in,
                              void* d_out, int out_size, void* d_ws, size_t ws_size,
                              hipStream_t stream) {
  const float* x   = (const float*)d_in[0];
  const float* Wsa = (const float*)d_in[1];
  const float* EF  = (const float*)d_in[2];
  const float* t2  = (const float*)d_in[3];
  const float* Wsc = (const float*)d_in[4];
  const float* tsc = (const float*)d_in[5];
  float* out = (float*)d_out;
  (void)in_sizes; (void)n_in; (void)out_size; (void)ws_size;

  char* ws = (char*)d_ws;
  size_t off = 0;
  auto take = [&](size_t bytes) -> char* {
    char* p = ws + off;
    off += (bytes + 255) & ~(size_t)255;
    return p;
  };
  _Float16* XH   = (_Float16*)take((size_t)Bn * Nn * Cc * 2);
  _Float16* QSA  = (_Float16*)take((size_t)Bn * Nn * Cc * 2);
  _Float16* KSAT = (_Float16*)take((size_t)Bn * Nn * Cc * 2);
  _Float16* VSAT = (_Float16*)take((size_t)Bn * Nn * Cc * 2);
  _Float16* QCT  = (_Float16*)take((size_t)Bn * Nn * Cc * 2);
  _Float16* KCT  = (_Float16*)take((size_t)Bn * Nn * Cc * 2);
  _Float16* VC   = (_Float16*)take((size_t)Bn * Nn * Cc * 2);
  _Float16* WSAH = (_Float16*)take((size_t)4 * Cc * Cc * 2);
  _Float16* WSCH = (_Float16*)take((size_t)4 * Cc * Cc * 2);
  _Float16* EFT  = (_Float16*)take((size_t)Nn * Pp * 2);
  _Float16* KPT  = (_Float16*)take((size_t)Bn * Hh * Pp * Dd * 2);
  _Float16* VP   = (_Float16*)take((size_t)Bn * Hh * Dd * Pp * 2);
  float*    ACA  = (float*)take((size_t)Bn * Hh * Dd * Dd * 4);
  _Float16* ACAH = (_Float16*)take((size_t)Bn * Hh * Dd * Dd * 2);
  float*    IQ   = (float*)take((size_t)Bn * Hh * Dd * 4);
  float*    IQC  = (float*)take((size_t)Bn * Hh * Dd * 4);
  float*    IKC  = (float*)take((size_t)Bn * Hh * Dd * 4);

  dim3 tb(32, 8);
  // xf[b][n][c] = x[b][c][n] (f16)
  k_transpose_f32_f16<<<dim3(Cc / 32, Nn / 32, Bn), tb, 0, stream>>>(
      XH, x, Cc, Nn, (long)Cc * Nn, (long)Nn * Cc);
  // EF^T[p][n]
  k_transpose_f32_f16<<<dim3(Nn / 32, Pp / 32, 1), tb, 0, stream>>>(
      EFT, EF, Nn, Pp, 0, 0);
  k_cvt_f16<<<(4 * Cc * Cc) / 256, 256, 0, stream>>>(WSAH, Wsa, 4 * Cc * Cc);
  k_cvt_f16<<<(4 * Cc * Cc) / 256, 256, 0, stream>>>(WSCH, Wsc, 4 * Cc * Cc);

  k_qkv_gemm<<<dim3(Nn / 128, 12, Bn * 2), 256, 0, stream>>>(
      XH, WSAH, WSCH, QSA, KSAT, VSAT, QCT, KCT, VC);

  k_norm_nd<<<Bn * Hh, 256, 0, stream>>>(QSA, IQ);
  k_norm_dn<<<dim3(Bn * Hh, 2), 256, 0, stream>>>(QCT, KCT, IQC, IKC);

  k_proj<<<dim3(Bn * Hh, 2), 256, 0, stream>>>(KSAT, VSAT, EFT, KPT, VP);
  k_ca_gemm<<<Bn * Hh, 128, 0, stream>>>(QCT, KCT, ACA);
  k_ca_softmax<<<Bn * Hh, 32, 0, stream>>>(ACA, IQC, IKC, tsc, ACAH);

  k_spatial<<<dim3(Bn * Hh, Nn / 128), 256, 0, stream>>>(QSA, KPT, VP, IQ, t2, out);
  k_channel<<<dim3(Bn * Hh, Nn / 64), 256, 0, stream>>>(ACAH, VC, out);
}
